// DynamicChannelMixer_49727131353105
// MI455X (gfx1250) — matrix operand order repaired and checked by reading, not compiled, not run on hardware
//
#include <hip/hip_runtime.h>
#include <math.h>

// Shapes (fixed by the reference)
#define Bn   16
#define Cc   64
#define Nn   256
#define Ee   64
#define Dd   256
#define NQ   8
#define FD   32
#define Hh   4
#define DH   64
#define BN   (Bn * Nn)           // 4096
#define KIN  (Ee + FD)           // 96

typedef __attribute__((ext_vector_type(16))) _Float16 v16h;
typedef __attribute__((ext_vector_type(8)))  float    v8f;

// ---------------- workspace layout (bytes, 256-aligned) ----------------
#define OFF_WC16     (0)                      // 24576 f16  -> 49152 B
#define OFF_WK16     (49152)                  // 65536 f16  (contiguous with WV)
#define OFF_WV16     (180224)                 // 65536 f16
#define OFF_WO16     (311296)                 // 65536 f16
#define OFF_WOUT16   (442368)                 // 524288 f16 -> 1 MB
#define OFF_Q        (1490944)                // 2048 f32 (q_scaled, bias+scale folded)
#define OFF_CF16     (1499136)                // 1024*32 f16
#define OFF_AOUT16   (1564672)                // 4096*2048 f16 -> 16 MB
#define OFF_ATTNP    (18341888)               // 4096*512 f32 -> 8 MB

// ---------------- fragment loaders (layouts per CDNA5 ISA 7.12.2) ------
__device__ __forceinline__ v16h load_a_frag(const _Float16* base, int row,
                                            int rowStride, int k0, int laneHalf) {
    const _Float16* p = base + row * rowStride + k0 + laneHalf * 8;
    union { float4 f[2]; v16h h; } u;
    u.f[0] = *(const float4*)(p);
    u.f[1] = *(const float4*)(p + 16);
    return u.h;
}
__device__ __forceinline__ v16h load_b_frag(const _Float16* base, int col,
                                            int rowStride, int k0, int laneHalf) {
    const _Float16* p = base + col * rowStride + k0 + laneHalf * 8;
    union { float4 f[2]; v16h h; } u;
    u.f[0] = *(const float4*)(p);
    u.f[1] = *(const float4*)(p + 16);
    return u.h;
}
#define WMMA(a, b, c) __builtin_amdgcn_wmma_f32_16x16x32_f16(false, (a), false, (b), (short)0, (c), false, false)

// ---------------- prologue: fp32 -> f16 weight conversion --------------
__global__ void cvt_f16_kernel(const float* __restrict__ src, _Float16* __restrict__ dst, int n) {
    int i = blockIdx.x * blockDim.x + threadIdx.x;
    if (i < n) dst[i] = (_Float16)src[i];
}

// q_scaled = (queries @ Wq.T + bq) / sqrt(dh);  cf = [sin, cos](coords @ B_mat)
__global__ void precompute_kernel(const float* __restrict__ coords,
                                  const float* __restrict__ Bmat,
                                  const float* __restrict__ queries,
                                  const float* __restrict__ Win,
                                  const float* __restrict__ bin,
                                  float* __restrict__ qout,
                                  _Float16* __restrict__ cf16) {
    int tid = blockIdx.x * blockDim.x + threadIdx.x;
    if (tid < NQ * Dd) {
        int q = tid >> 8, dc = tid & 255;
        float acc = bin[dc];
        for (int e = 0; e < Dd; ++e) acc += queries[q * Dd + e] * Win[dc * Dd + e];
        qout[tid] = acc * 0.125f;   // 1/sqrt(64)
    } else {
        int t2 = tid - NQ * Dd;
        if (t2 < Bn * Cc * (FD / 2)) {
            int row = t2 >> 4, j = t2 & 15;
            float p = coords[row * 3 + 0] * Bmat[j]
                    + coords[row * 3 + 1] * Bmat[16 + j]
                    + coords[row * 3 + 2] * Bmat[32 + j];
            cf16[row * FD + j]      = (_Float16)sinf(p);
            cf16[row * FD + 16 + j] = (_Float16)cosf(p);
        }
    }
}

// ---------------- main fused kernel: one block per (b, n) --------------
__global__ void mixer_main_kernel(const float* __restrict__ patches,
                                  const unsigned char* __restrict__ mask,
                                  const float* __restrict__ bc,
                                  const float* __restrict__ bin,
                                  const float* __restrict__ bo,
                                  const _Float16* __restrict__ Wc16,
                                  const _Float16* __restrict__ Wk16,
                                  const _Float16* __restrict__ Wv16,
                                  const _Float16* __restrict__ Wo16,
                                  const float* __restrict__ qscaled,
                                  const _Float16* __restrict__ cf16,
                                  _Float16* __restrict__ attnout16,
                                  float* __restrict__ attnp) {
    extern __shared__ char smem[];
    _Float16* sX  = (_Float16*)(smem);             // [64][96]   12288 B
    _Float16* sKV = (_Float16*)(smem + 12288);     // [64][256]  32768 B
    _Float16* sK  = (_Float16*)(smem + 45056);     // [64][256]
    _Float16* sV  = (_Float16*)(smem + 77824);     // [64][256]
    float*    sS  = (float*)   (smem + 110592);    // [4][8][64]  8192 B
    _Float16* sCtx= (_Float16*)(smem + 118784);    // [16][256]   8192 B

    const int bn   = blockIdx.x;
    const int b    = bn / Nn;
    const int nt_t = bn % Nn;
    const int tid  = threadIdx.x;
    const int wave = tid >> 5;
    const int lane = tid & 31;
    const int l16  = lane & 15;
    const int lh   = lane >> 4;

    // ---- stage X = [patches_row | cf] as f16 in LDS ----
    for (int idx = tid; idx < Cc * KIN; idx += 256) {
        int c = idx / KIN, k = idx % KIN;
        if (k < Ee) {
            sX[c * KIN + k] = (_Float16)patches[(((b * Cc + c) * Nn) + nt_t) * Ee + k];
        } else {
            sX[c * KIN + k] = cf16[(b * Cc + c) * FD + (k - Ee)];
        }
    }
    __syncthreads();

    // ---- Phase 1: kv = X @ Wc^T + bc   (64x96 @ 96x256) ----
    {
        const int mt = wave & 3;
        const int ntb = (wave >> 2) * 8;
        for (int p = 0; p < 4; ++p) {
            int nt0 = ntb + 2 * p;
            v8f acc0 = {}, acc1 = {};
            #pragma unroll
            for (int kt = 0; kt < 3; ++kt) {
                int k0 = kt * 32;
                v16h a  = load_a_frag(sX, mt * 16 + l16, KIN, k0, lh);
                v16h b0 = load_b_frag(Wc16, nt0 * 16 + l16, KIN, k0, lh);
                v16h b1 = load_b_frag(Wc16, (nt0 + 1) * 16 + l16, KIN, k0, lh);
                acc0 = WMMA(a, b0, acc0);
                acc1 = WMMA(a, b1, acc1);
            }
            #pragma unroll
            for (int t = 0; t < 2; ++t) {
                int ncol = (nt0 + t) * 16 + l16;
                float bias = bc[ncol];
                v8f acc = t ? acc1 : acc0;
                #pragma unroll
                for (int r = 0; r < 8; ++r) {
                    int m = mt * 16 + lh * 8 + r;
                    sKV[m * Dd + ncol] = (_Float16)(acc[r] + bias);
                }
            }
        }
    }
    __syncthreads();

    // ---- Phase 2: k = kv @ Wk^T + bk ; v = kv @ Wv^T + bv ----
    for (int pass = 0; pass < 2; ++pass) {
        const _Float16* W = pass ? Wv16 : Wk16;
        const float* bias_p = bin + Dd + pass * Dd;
        _Float16* dst = pass ? sV : sK;
        const int mt = wave & 3;
        const int ntb = (wave >> 2) * 8;
        for (int p = 0; p < 4; ++p) {
            int nt0 = ntb + 2 * p;
            v8f acc0 = {}, acc1 = {};
            #pragma unroll
            for (int kt = 0; kt < 8; ++kt) {
                int k0 = kt * 32;
                v16h a  = load_a_frag(sKV, mt * 16 + l16, Dd, k0, lh);
                v16h b0 = load_b_frag(W, nt0 * 16 + l16, Dd, k0, lh);
                v16h b1 = load_b_frag(W, (nt0 + 1) * 16 + l16, Dd, k0, lh);
                acc0 = WMMA(a, b0, acc0);
                acc1 = WMMA(a, b1, acc1);
            }
            #pragma unroll
            for (int t = 0; t < 2; ++t) {
                int ncol = (nt0 + t) * 16 + l16;
                float bias = bias_p[ncol];
                v8f acc = t ? acc1 : acc0;
                #pragma unroll
                for (int r = 0; r < 8; ++r) {
                    int m = mt * 16 + lh * 8 + r;
                    dst[m * Dd + ncol] = (_Float16)(acc[r] + bias);
                }
            }
        }
        __syncthreads();
    }

    // ---- Phase 3: scores -> softmax -> ctx (small, VALU) ----
    for (int idx = tid; idx < Hh * NQ * Cc; idx += 256) {
        int h = idx >> 9, q = (idx >> 6) & 7, c = idx & 63;
        const float*    qr = qscaled + q * Dd + h * DH;
        const _Float16* kr = sK + c * Dd + h * DH;
        float s = 0.f;
        #pragma unroll 8
        for (int dd = 0; dd < DH; ++dd) s += qr[dd] * (float)kr[dd];
        if (mask[b * Cc + c]) s = -3.4e38f;
        sS[idx] = s;
    }
    __syncthreads();
    if (tid < Hh * NQ) {          // one (h,q) row per thread
        float* row = sS + tid * Cc;
        float mx = -3.4e38f;
        for (int c = 0; c < Cc; ++c) mx = fmaxf(mx, row[c]);
        float sum = 0.f;
        for (int c = 0; c < Cc; ++c) { float e = __expf(row[c] - mx); row[c] = e; sum += e; }
        float inv = 1.0f / sum;
        for (int c = 0; c < Cc; ++c) row[c] *= inv;
    }
    __syncthreads();
    // deterministic attn partial: mean over heads, summed over n later
    for (int idx = tid; idx < NQ * Cc; idx += 256) {
        int q = idx >> 6, c = idx & 63;
        float s = 0.f;
        #pragma unroll
        for (int h = 0; h < Hh; ++h) s += sS[((h << 3) + q) * Cc + c];
        attnp[bn * (NQ * Cc) + idx] = s * 0.25f;
    }
    // ctx[q][d] padded to 16 rows for WMMA-A
    for (int idx = tid; idx < 16 * Dd; idx += 256) {
        int q = idx >> 8, d = idx & 255;
        if (q < NQ) {
            int h = d >> 6;
            float s = 0.f;
            #pragma unroll 8
            for (int c = 0; c < Cc; ++c)
                s += sS[((h << 3) + q) * Cc + c] * (float)sV[c * Dd + d];
            sCtx[idx] = (_Float16)s;
        } else {
            sCtx[idx] = (_Float16)0.f;
        }
    }
    __syncthreads();

    // ---- Phase 4: attn_out = ctx @ Wo^T + bo  -> workspace (f16) ----
    {
        int nt0 = wave * 2;
        v8f acc0 = {}, acc1 = {};
        #pragma unroll
        for (int kt = 0; kt < 8; ++kt) {
            int k0 = kt * 32;
            v16h a  = load_a_frag(sCtx, l16, Dd, k0, lh);
            v16h b0 = load_b_frag(Wo16, nt0 * 16 + l16, Dd, k0, lh);
            v16h b1 = load_b_frag(Wo16, (nt0 + 1) * 16 + l16, Dd, k0, lh);
            acc0 = WMMA(a, b0, acc0);
            acc1 = WMMA(a, b1, acc1);
        }
        #pragma unroll
        for (int t = 0; t < 2; ++t) {
            int ncol = (nt0 + t) * 16 + l16;
            float bias = bo[ncol];
            v8f acc = t ? acc1 : acc0;
            if (lh == 0) {   // rows 0..7 are the valid nq rows
                #pragma unroll
                for (int r = 0; r < 8; ++r)
                    attnout16[(size_t)bn * (NQ * Dd) + r * Dd + ncol] = (_Float16)(acc[r] + bias);
            }
        }
    }
}

// ---------------- attn_weights: mean over n (deterministic) ------------
__global__ void attn_reduce_kernel(const float* __restrict__ attnp, float* __restrict__ out_attn) {
    int i = blockIdx.x * blockDim.x + threadIdx.x;   // 8192 total
    if (i < Bn * NQ * Cc) {
        int b = i >> 9, qc = i & 511;
        const float* p = attnp + (size_t)b * Nn * (NQ * Cc) + qc;
        float s = 0.f;
        for (int n = 0; n < Nn; ++n) s += p[n * (NQ * Cc)];
        out_attn[i] = s * (1.0f / (float)Nn);
    }
}

// ---------------- final: seq = attn_out_flat @ Wout^T + bout, then LN ---
__global__ void seq_gemm_ln_kernel(const _Float16* __restrict__ A,      // [4096][2048]
                                   const _Float16* __restrict__ Wout16, // [256][2048]
                                   const float* __restrict__ bout,
                                   const float* __restrict__ gamma,
                                   const float* __restrict__ beta,
                                   float* __restrict__ out) {
    extern __shared__ char smem[];
    float* sSeq = (float*)smem;                 // [16][256]  16384 B
    float* sMu  = (float*)(smem + 16384);       // [16]
    float* sRs  = (float*)(smem + 16448);       // [16]

    const int mtile = blockIdx.x;               // 256 M-tiles of 16 rows
    const int tid = threadIdx.x;
    const int wave = tid >> 5, lane = tid & 31;
    const int l16 = lane & 15, lh = lane >> 4;
    const int K = NQ * Dd;                      // 2048

    int nt0 = wave * 2;
    v8f acc0 = {}, acc1 = {};
    for (int kt = 0; kt < K / 32; ++kt) {
        int k0 = kt * 32;
        v16h a  = load_a_frag(A, mtile * 16 + l16, K, k0, lh);
        v16h b0 = load_b_frag(Wout16, nt0 * 16 + l16, K, k0, lh);
        v16h b1 = load_b_frag(Wout16, (nt0 + 1) * 16 + l16, K, k0, lh);
        acc0 = WMMA(a, b0, acc0);
        acc1 = WMMA(a, b1, acc1);
    }
    #pragma unroll
    for (int t = 0; t < 2; ++t) {
        int ncol = (nt0 + t) * 16 + l16;
        float bias = bout[ncol];
        v8f acc = t ? acc1 : acc0;
        #pragma unroll
        for (int r = 0; r < 8; ++r)
            sSeq[(lh * 8 + r) * Dd + ncol] = acc[r] + bias;
    }
    __syncthreads();
    if (tid < 16) {
        const float* row = sSeq + tid * Dd;
        float mu = 0.f;
        for (int c = 0; c < Dd; ++c) mu += row[c];
        mu *= (1.0f / Dd);
        float var = 0.f;
        for (int c = 0; c < Dd; ++c) { float d = row[c] - mu; var += d * d; }
        var *= (1.0f / Dd);
        sMu[tid] = mu;
        sRs[tid] = rsqrtf(var + 1e-5f);
    }
    __syncthreads();
    for (int idx = tid; idx < 16 * Dd; idx += 256) {
        int row = idx >> 8, col = idx & 255;
        out[(size_t)(mtile * 16 + row) * Dd + col] =
            (sSeq[idx] - sMu[row]) * sRs[row] * gamma[col] + beta[col];
    }
}

// ------------------------------- launch --------------------------------
extern "C" void kernel_launch(void* const* d_in, const int* in_sizes, int n_in,
                              void* d_out, int out_size, void* d_ws, size_t ws_size,
                              hipStream_t stream) {
    const float* patches = (const float*)d_in[0];
    const float* coords  = (const float*)d_in[1];
    const unsigned char* mask = (const unsigned char*)d_in[2];
    const float* Bmat    = (const float*)d_in[3];
    const float* Wc      = (const float*)d_in[4];
    const float* bc      = (const float*)d_in[5];
    const float* queries = (const float*)d_in[6];
    const float* Win     = (const float*)d_in[7];
    const float* bin     = (const float*)d_in[8];
    const float* Wo      = (const float*)d_in[9];
    const float* bo      = (const float*)d_in[10];
    const float* Wout    = (const float*)d_in[11];
    const float* bout    = (const float*)d_in[12];
    const float* gamma   = (const float*)d_in[13];
    const float* beta    = (const float*)d_in[14];
    float* out = (float*)d_out;
    char* ws = (char*)d_ws;

    _Float16* wc16   = (_Float16*)(ws + OFF_WC16);
    _Float16* wk16   = (_Float16*)(ws + OFF_WK16);
    _Float16* wv16   = (_Float16*)(ws + OFF_WV16);
    _Float16* wo16   = (_Float16*)(ws + OFF_WO16);
    _Float16* wout16 = (_Float16*)(ws + OFF_WOUT16);
    float*    qbuf   = (float*)   (ws + OFF_Q);
    _Float16* cf16   = (_Float16*)(ws + OFF_CF16);
    _Float16* aout16 = (_Float16*)(ws + OFF_AOUT16);
    float*    attnp  = (float*)   (ws + OFF_ATTNP);

    cvt_f16_kernel<<<(Dd * KIN + 255) / 256, 256, 0, stream>>>(Wc, wc16, Dd * KIN);
    // Wk and Wv are contiguous rows [256..768) of Win -> contiguous f16 region
    cvt_f16_kernel<<<(2 * Dd * Dd + 255) / 256, 256, 0, stream>>>(Win + Dd * Dd, wk16, 2 * Dd * Dd);
    cvt_f16_kernel<<<(Dd * Dd + 255) / 256, 256, 0, stream>>>(Wo, wo16, Dd * Dd);
    cvt_f16_kernel<<<(Dd * NQ * Dd + 255) / 256, 256, 0, stream>>>(Wout, wout16, Dd * NQ * Dd);

    precompute_kernel<<<(NQ * Dd + Bn * Cc * (FD / 2) + 255) / 256, 256, 0, stream>>>(
        coords, Bmat, queries, Win, bin, qbuf, cf16);

    mixer_main_kernel<<<BN, 256, 126976, stream>>>(
        patches, mask, bc, bin, bo, wc16, wk16, wv16, wo16, qbuf, cf16, aout16, attnp);

    attn_reduce_kernel<<<(Bn * NQ * Cc + 255) / 256, 256, 0, stream>>>(attnp, out + (size_t)BN * Dd);

    seq_gemm_ln_kernel<<<BN / 16, 256, 16512, stream>>>(aout16, wout16, bout, gamma, beta, out);
}
